// Gate_40037685133871
// MI455X (gfx1250) — matrix operand order repaired and checked by reading, not compiled, not run on hardware
//
#include <hip/hip_runtime.h>
#include <math.h>

typedef __bf16 bf16_t;
typedef __attribute__((ext_vector_type(16))) __bf16 v16bf;
typedef __attribute__((ext_vector_type(8)))  __bf16 v8bf;
typedef __attribute__((ext_vector_type(8)))  float  v8f;
typedef __attribute__((ext_vector_type(4)))  float  v4f;

#define SEQ    32
#define NEXP   9
#define NFREQ  16
#define BATCH  64
#define CDIM   512            // 4*D
#define KDIM   32768          // D*H*W
#define MDIM   2048           // B*T

#define TM 128
#define TN 64
#define TK 64
#define LDSTR 72              // padded LDS row stride (elements) = 144B: aligned + conflict-free
#define SPLITK 4
#define KSEG (KDIM / SPLITK)  // 8192

// ---------------------------------------------------------------------------
// Kernel 0: permute+convert conv_w (N,K) fp32 with k=d*256+hw  ->  bf16 Bt[n][k']
// where k' = hw*128 + d matches x's native memory order.
// ---------------------------------------------------------------------------
__global__ __launch_bounds__(256)
void convert_w_kernel(const float* __restrict__ w, bf16_t* __restrict__ Bt) {
    size_t i = (size_t)blockIdx.x * 256 + threadIdx.x;   // over 512*32768
    int hw = (int)(i & 255);
    int d  = (int)((i >> 8) & 127);
    size_t n = i >> 15;
    Bt[n * (size_t)KDIM + (size_t)hw * 128 + d] = (bf16_t)w[i];
}

// ---------------------------------------------------------------------------
// Kernel 1: bf16 WMMA GEMM, double-buffered pipeline.
//   A tile (fp32 -> bf16) staged through VGPRs, drained with DS stores;
//   B tile (already bf16) moved with CDNA5 async global->LDS loads (ASYNCcnt).
// ---------------------------------------------------------------------------
__global__ __launch_bounds__(256)
void gemm_feat_kernel(const float* __restrict__ X, const bf16_t* __restrict__ Bt,
                      float* __restrict__ featp) {
    __shared__ __align__(16) bf16_t lA[2][TM * LDSTR];   // 2 x 18 KB
    __shared__ __align__(16) bf16_t lB[2][TN * LDSTR];   // 2 x  9 KB

    const int tid  = threadIdx.x;
    const int lane = tid & 31;
    const int wave = tid >> 5;
    const int wm   = wave & 3;      // wave tile row 0..3
    const int wn   = wave >> 2;     // wave tile col 0..1
    const int half = lane >> 4;     // lane half (wave32)
    const int r    = lane & 15;

    const size_t m0   = (size_t)blockIdx.y * TM;
    const size_t n0   = (size_t)blockIdx.x * TN;
    const size_t kbeg = (size_t)blockIdx.z * KSEG;

    // cooperative loader mapping
    const int arow = tid >> 1;            // 0..127
    const int acol = (tid & 1) * 32;      // 0 | 32   (32 floats per thread)
    const int brow = tid >> 2;            // 0..63
    const int bcol = (tid & 3) * 16;      // 0,16,32,48 (2 x b128 per thread)

    const float*  aSrc = X  + (m0 + arow) * (size_t)KDIM + acol;
    const bf16_t* bSrc = Bt + (n0 + brow) * (size_t)KDIM + bcol;
    // keep a single AS(3)-derived base; select buffer by index arithmetic so
    // the compiler emits ds_store (not flat_store) for the drain.
    bf16_t* aBase = &lA[0][0] + arow * LDSTR + acol;
    // low 32 bits of a generic __shared__ pointer == LDS byte address
    unsigned bLds[2] = { (unsigned)(size_t)&lB[0][brow * LDSTR + bcol],
                         (unsigned)(size_t)&lB[1][brow * LDSTR + bcol] };

    v8f acc[2][2] = {};
    v4f fr[8];

    // ---- prologue: tile 0 ----
    {
        unsigned long long g = (unsigned long long)(bSrc + kbeg);
        asm volatile("global_load_async_to_lds_b128 %0, %1, off"
                     :: "v"(bLds[0]), "v"(g) : "memory");
        asm volatile("global_load_async_to_lds_b128 %0, %1, off"
                     :: "v"(bLds[0] + 16u), "v"(g + 16ull) : "memory");
        const float* s = aSrc + kbeg;
#pragma unroll
        for (int i = 0; i < 8; ++i) fr[i] = *(const v4f*)(s + 4 * i);
        bf16_t* d = aBase;
#pragma unroll
        for (int i = 0; i < 8; ++i)
#pragma unroll
            for (int j = 0; j < 4; ++j) d[4 * i + j] = (bf16_t)fr[i][j];
        asm volatile("s_wait_asynccnt 0" ::: "memory");
    }
    __syncthreads();

    const int NIT = KSEG / TK;   // 128
    for (int it = 0; it < NIT; ++it) {
        const int  cur  = it & 1;
        const int  nxt  = cur ^ 1;
        const bool more = (it + 1) < NIT;

        // ---- issue next tile's loads (async B -> LDS, A -> VGPRs) ----
        if (more) {
            const size_t kn = kbeg + (size_t)(it + 1) * TK;
            unsigned long long g = (unsigned long long)(bSrc + kn);
            asm volatile("global_load_async_to_lds_b128 %0, %1, off"
                         :: "v"(bLds[nxt]), "v"(g) : "memory");
            asm volatile("global_load_async_to_lds_b128 %0, %1, off"
                         :: "v"(bLds[nxt] + 16u), "v"(g + 16ull) : "memory");
            const float* s = aSrc + kn;
#pragma unroll
            for (int i = 0; i < 8; ++i) fr[i] = *(const v4f*)(s + 4 * i);
        }

        // ---- compute on current buffer: 2 K-steps x (2x2) WMMA ----
        const bf16_t* As = lA[cur];
        const bf16_t* Bs = lB[cur];
#pragma unroll
        for (int kk = 0; kk < 2; ++kk) {
            v16bf af[2], bfm[2];
#pragma unroll
            for (int f = 0; f < 2; ++f) {
                // A 16x32 frag: half 0 -> K{0..7,16..23}, half 1 -> K{8..15,24..31}
                const bf16_t* ap = &As[(wm * 32 + f * 16 + r) * LDSTR + kk * 32];
                v8bf alo = *(const v8bf*)(ap + half * 8);
                v8bf ahi = *(const v8bf*)(ap + 16 + half * 8);
#pragma unroll
                for (int i = 0; i < 8; ++i) { af[f][i] = alo[i]; af[f][8 + i] = ahi[i]; }
                // B 32x16 frag: half 0 -> K{0..7,16..23}, half 1 -> K{8..15,24..31} (col = r)
                const bf16_t* bp = &Bs[(wn * 32 + f * 16 + r) * LDSTR + kk * 32];
                v8bf blo = *(const v8bf*)(bp + half * 8);
                v8bf bhi = *(const v8bf*)(bp + 16 + half * 8);
#pragma unroll
                for (int i = 0; i < 8; ++i) { bfm[f][i] = blo[i]; bfm[f][8 + i] = bhi[i]; }
            }
#pragma unroll
            for (int i = 0; i < 2; ++i)
#pragma unroll
                for (int j = 0; j < 2; ++j)
                    acc[i][j] = __builtin_amdgcn_wmma_f32_16x16x32_bf16(
                        false, af[i], false, bfm[j], (short)0, acc[i][j], false, false);
        }

        // ---- drain staged A into next buffer; wait this wave's async B ----
        if (more) {
            bf16_t* d = aBase + nxt * (TM * LDSTR);
#pragma unroll
            for (int i = 0; i < 8; ++i)
#pragma unroll
                for (int j = 0; j < 4; ++j) d[4 * i + j] = (bf16_t)fr[i][j];
            asm volatile("s_wait_asynccnt 0" ::: "memory");
        }
        __syncthreads();
    }

    // ---- epilogue: D VGPR q holds row (q + half*8), col r ----
    float* out = featp + (size_t)blockIdx.z * MDIM * CDIM;
#pragma unroll
    for (int i = 0; i < 2; ++i) {
        size_t mbase = m0 + wm * 32 + i * 16 + half * 8;
#pragma unroll
        for (int j = 0; j < 2; ++j) {
            size_t ncol = n0 + wn * 32 + j * 16 + r;
#pragma unroll
            for (int q = 0; q < 8; ++q)
                out[(mbase + q) * CDIM + ncol] = acc[i][j][q];
        }
    }
}

// ---------------------------------------------------------------------------
// Kernel 2: deterministic split-K reduction + bias
// ---------------------------------------------------------------------------
__global__ __launch_bounds__(256)
void reduce_feat_kernel(const float* __restrict__ featp, const float* __restrict__ bias,
                        float* __restrict__ feat) {
    size_t i = (size_t)blockIdx.x * 256 + threadIdx.x;   // < MDIM*CDIM
    int n = (int)(i & (CDIM - 1));
    float s = bias[n];
#pragma unroll
    for (int z = 0; z < SPLITK; ++z)
        s += featp[(size_t)z * MDIM * CDIM + i];
    feat[i] = s;
}

// ---------------------------------------------------------------------------
// Kernel 3: rFFT magnitudes over T=32, drop DC, mean over 512 channels (ortho)
// ---------------------------------------------------------------------------
__global__ __launch_bounds__(256)
void amp_kernel(const float* __restrict__ feat, float* __restrict__ amp) {
    __shared__ float ct[NFREQ * SEQ];
    __shared__ float st[NFREQ * SEQ];
    __shared__ float red[256 * NFREQ];
    const int b = blockIdx.x;
    const int tid = threadIdx.x;

    for (int i = tid; i < NFREQ * SEQ; i += 256) {
        int k = i / SEQ + 1, t = i % SEQ;           // k = 1..16 (DC dropped)
        float ang = -6.283185307179586f * (float)(k * t) / 32.0f;
        ct[i] = cosf(ang);
        st[i] = sinf(ang);
    }
    __syncthreads();

    float local[NFREQ];
#pragma unroll
    for (int k = 0; k < NFREQ; ++k) local[k] = 0.0f;

    for (int c = tid; c < CDIM; c += 256) {
        float re[NFREQ], im[NFREQ];
#pragma unroll
        for (int k = 0; k < NFREQ; ++k) { re[k] = 0.0f; im[k] = 0.0f; }
        for (int t = 0; t < SEQ; ++t) {
            float f = feat[((size_t)b * SEQ + t) * CDIM + c];
#pragma unroll
            for (int k = 0; k < NFREQ; ++k) {
                re[k] = fmaf(f, ct[k * SEQ + t], re[k]);
                im[k] = fmaf(f, st[k * SEQ + t], im[k]);
            }
        }
#pragma unroll
        for (int k = 0; k < NFREQ; ++k)
            local[k] += sqrtf(re[k] * re[k] + im[k] * im[k]);
    }
#pragma unroll
    for (int k = 0; k < NFREQ; ++k) red[tid * NFREQ + k] = local[k];
    __syncthreads();

    if (tid < NFREQ) {
        float s = 0.0f;
        for (int t = 0; t < 256; ++t) s += red[t * NFREQ + tid];
        amp[b * NFREQ + tid] = s * (1.0f / (float)CDIM) * 0.17677669529663687f;
    }
}

// ---------------------------------------------------------------------------
// Kernel 4: noisy-top-k gating + load (one 64-thread block, thread = batch row)
// ---------------------------------------------------------------------------
__global__ __launch_bounds__(64)
void gate_kernel(const float* __restrict__ amp, const float* __restrict__ w_gate,
                 const float* __restrict__ w_noise, const float* __restrict__ noise,
                 const int* __restrict__ training, float* __restrict__ out) {
    __shared__ float probs[BATCH * NEXP];
    const int b = threadIdx.x;
    const int train = training[0];

    float a[NFREQ];
#pragma unroll
    for (int i = 0; i < NFREQ; ++i) a[i] = amp[b * NFREQ + i];

    float cl[NEXP], sd[NEXP], lg[NEXP];
#pragma unroll
    for (int e = 0; e < NEXP; ++e) {
        float g = 0.0f, nn = 0.0f;
#pragma unroll
        for (int i = 0; i < NFREQ; ++i) {
            g  = fmaf(a[i], w_gate[i * NEXP + e], g);
            nn = fmaf(a[i], w_noise[i * NEXP + e], nn);
        }
        cl[e] = g;
        float sp = fmaxf(nn, 0.0f) + log1pf(expf(-fabsf(nn)));   // stable softplus
        sd[e] = sp + 0.01f;
        lg[e] = train ? (g + noise[b * NEXP + e] * sd[e]) : g;
    }

    // top-3 (values descending; indices for top-2)
    float t0 = -1e30f, t1 = -1e30f, t2 = -1e30f;
    int i0 = 0, i1 = 0;
#pragma unroll
    for (int e = 0; e < NEXP; ++e) {
        float v = lg[e];
        if (v > t0)      { t2 = t1; t1 = t0; i1 = i0; t0 = v; i0 = e; }
        else if (v > t1) { t2 = t1; t1 = v;  i1 = e; }
        else if (v > t2) { t2 = v; }
    }

    // softmax over top-2, scatter
    float e1 = expf(t1 - t0);
    float inv = 1.0f / (1.0f + e1);
#pragma unroll
    for (int e = 0; e < NEXP; ++e) out[b * NEXP + e] = 0.0f;
    out[b * NEXP + i0] = inv;
    out[b * NEXP + i1] = e1 * inv;

    // load
#pragma unroll
    for (int e = 0; e < NEXP; ++e) {
        float p;
        if (train) {
            bool  is_in = lg[e] > t2;                 // thr_in = 3rd largest
            float thr   = is_in ? t2 : t1;            // thr_out = 2nd largest
            float z = (cl[e] - thr) / sd[e];
            p = 0.5f * erfcf(-z * 0.70710678118654752f);   // Phi(z)
        } else {
            p = (e == i0 || e == i1) ? 1.0f : 0.0f;
        }
        probs[b * NEXP + e] = p;
    }
    __syncthreads();
    if (b < NEXP) {
        float s = 0.0f;
        for (int bb = 0; bb < BATCH; ++bb) s += probs[bb * NEXP + b];
        out[BATCH * NEXP + b] = s;
    }
}

// ---------------------------------------------------------------------------
extern "C" void kernel_launch(void* const* d_in, const int* in_sizes, int n_in,
                              void* d_out, int out_size, void* d_ws, size_t ws_size,
                              hipStream_t stream) {
    (void)in_sizes; (void)n_in; (void)out_size; (void)ws_size;
    const float* x        = (const float*)d_in[0];
    const float* conv_w   = (const float*)d_in[1];
    const float* conv_b   = (const float*)d_in[2];
    const float* w_gate   = (const float*)d_in[3];
    const float* w_noise  = (const float*)d_in[4];
    const float* noise    = (const float*)d_in[5];
    const int*   training = (const int*)d_in[6];
    float* out = (float*)d_out;

    char* ws = (char*)d_ws;
    bf16_t* Bt    = (bf16_t*)ws;                                          // 33.55 MB
    float*  featp = (float*)(ws + (size_t)CDIM * KDIM * sizeof(bf16_t));  // 16.78 MB
    float*  feat  = featp + (size_t)SPLITK * MDIM * CDIM;                 //  4.19 MB
    float*  amp   = feat + (size_t)MDIM * CDIM;                           //  4 KB

    convert_w_kernel<<<((size_t)CDIM * KDIM) / 256, 256, 0, stream>>>(conv_w, Bt);

    dim3 grid(CDIM / TN, MDIM / TM, SPLITK);   // (8, 16, 4)
    gemm_feat_kernel<<<grid, 256, 0, stream>>>(x, Bt, featp);

    reduce_feat_kernel<<<((size_t)MDIM * CDIM) / 256, 256, 0, stream>>>(featp, conv_b, feat);

    amp_kernel<<<BATCH, 256, 0, stream>>>(feat, amp);

    gate_kernel<<<1, 64, 0, stream>>>(amp, w_gate, w_noise, noise, training, out);
}
